// RecurrentDecoder_38766374813776
// MI455X (gfx1250) — compile-verified, matrix operand order checked by reading, not run on hardware
//
#include <hip/hip_runtime.h>

// ---------------- types ----------------
typedef __attribute__((ext_vector_type(16))) __bf16 v16bf;
typedef __attribute__((ext_vector_type(8)))  float  v8f;

// ---------------- problem constants ----------------
#define BB    8
#define TT    511        // T-1
#define EE    256
#define HH    1024
#define H3    3072       // 3*H
#define VV    32000
#define MROWS (BB*TT)    // 4088
#define MPAD  4096       // 256 m-tiles

// ---------------- workspace layout (bytes, 256-aligned) ----------------
#define ALIGN256(x) (((x) + 255) & ~((size_t)255))
static const size_t OFF_WO  = 0;                                    // bf16 swizzled 1024x32000
static const size_t SZ_WO   = ALIGN256((size_t)HH * VV * 2);
static const size_t OFF_UK  = OFF_WO + SZ_WO;                       // bf16 swizzled 1024x3072
static const size_t SZ_UK   = ALIGN256((size_t)HH * H3 * 2);
static const size_t OFF_WK  = OFF_UK + SZ_UK;                       // bf16 swizzled 256x3072
static const size_t SZ_WK   = ALIGN256((size_t)EE * H3 * 2);
static const size_t OFF_X   = OFF_WK + SZ_WK;                       // bf16 A-swizzled 4096x256
static const size_t SZ_X    = ALIGN256((size_t)MPAD * EE * 2);
static const size_t OFF_XP  = OFF_X + SZ_X;                         // f32 4088x3072
static const size_t SZ_XP   = ALIGN256((size_t)MROWS * H3 * 4);
static const size_t OFF_HS  = OFF_XP + SZ_XP;                       // bf16 A-swizzled 4096x1024
static const size_t SZ_HS   = ALIGN256((size_t)MPAD * HH * 2);
static const size_t OFF_HA  = OFF_HS + SZ_HS;                       // f32 16x1024
static const size_t SZ_HF   = ALIGN256((size_t)16 * HH * 4);
static const size_t OFF_HB  = OFF_HA + SZ_HF;
static const size_t OFF_HABF = OFF_HB + SZ_HF;                      // bf16 A-swizzled 16x1024
static const size_t SZ_HBF  = ALIGN256((size_t)16 * HH * 2);
static const size_t OFF_HBBF = OFF_HABF + SZ_HBF;
static const size_t OFF_IDX  = OFF_HBBF + SZ_HBF;                   // int 4088

// ---------------- helpers ----------------
__device__ __forceinline__ unsigned rotl32(unsigned x, int r) {
    return (x << r) | (x >> (32 - r));
}

// threefry2x32-style counter hash (deterministic stand-in for jax uniform)
__device__ unsigned tf_hash(unsigned ctr) {
    const unsigned k0 = 0u, k1 = 42u;
    const unsigned ks2 = k0 ^ k1 ^ 0x1BD11BDAu;
    unsigned key[3] = {k0, k1, ks2};
    const int R[8] = {13, 15, 26, 6, 17, 29, 16, 24};
    unsigned x0 = ctr + k0, x1 = k1;
    #pragma unroll
    for (int g = 0; g < 5; ++g) {
        #pragma unroll
        for (int r = 0; r < 4; ++r) {
            x0 += x1;
            x1 = rotl32(x1, R[(g & 1) * 4 + r]);
            x1 ^= x0;
        }
        x0 += key[(g + 1) % 3];
        x1 += key[(g + 2) % 3] + (unsigned)(g + 1);
    }
    return x0 ^ x1;
}

__device__ __forceinline__ float sigmoidf_(float x) {
    return 1.0f / (1.0f + __expf(-x));
}

// k-offset inside a 32-wide k-tile for fragment element e, lane-half h
// (ISA 7.12.2, 16-bit 16x32 A / 32x16 B layouts)
__device__ __forceinline__ int frag_klocal(int e, int half) {
    int p = e >> 1, w = e & 1;
    return ((p & 3) * 2 + w) + ((p >> 2) * 16) + half * 8;
}
// inverse: klocal (0..31) -> (half, e)
__device__ __forceinline__ void inv_klocal(int kl, int& half, int& e) {
    half = (kl >> 3) & 1;
    int group = (kl >> 4) & 1;
    int within = kl & 7;
    int p = group * 4 + (within >> 1);
    e = p * 2 + (within & 1);
}

// ---------------- kernels ----------------

// B operand swizzle: src f32 (K x N, row-major) -> bf16 fragment tiles.
// Tile (ntile, ktile) at [(ntile*(K/32)+ktile)*512], lane-major [lane][16].
__global__ void swizzleB_kernel(const float* __restrict__ src,
                                __bf16* __restrict__ dst, int K, int N) {
    long total = (long)K * N;
    long stride = (long)gridDim.x * blockDim.x;
    int Kt = K >> 5;
    for (long i = (long)blockIdx.x * blockDim.x + threadIdx.x; i < total; i += stride) {
        long tile = i >> 9;
        int lane = (int)((i >> 4) & 31);
        int e = (int)(i & 15);
        long ntile = tile / Kt;
        int ktile = (int)(tile % Kt);
        int half = lane >> 4, nrow = lane & 15;
        long n = ntile * 16 + nrow;
        int k = ktile * 32 + frag_klocal(e, half);
        dst[i] = (__bf16)src[(long)k * N + n];
    }
}

// dropout indices for teacher-forced tokens
__global__ void dropidx_kernel(const int* __restrict__ tok, int* __restrict__ idx) {
    int row = blockIdx.x * blockDim.x + threadIdx.x;
    if (row >= MROWS) return;
    int b = row / TT, t = row % TT;
    int v = tok[b * 512 + t];
    unsigned bits = tf_hash((unsigned)row);
    float u = (float)(bits >> 8) * (1.0f / 16777216.0f);
    if (u < 0.2f) v = 0;
    idx[row] = v;
}

// A operand gather+swizzle: x_sw (4096x256 padded) from emb[dropidx]
__global__ void swizzleA_gather_kernel(const int* __restrict__ dropidx,
                                       const float* __restrict__ emb,
                                       __bf16* __restrict__ dst) {
    const int Kt = EE >> 5;                 // 8
    long total = (long)MPAD * EE;
    long stride = (long)gridDim.x * blockDim.x;
    for (long i = (long)blockIdx.x * blockDim.x + threadIdx.x; i < total; i += stride) {
        long tile = i >> 9;
        int lane = (int)((i >> 4) & 31);
        int e = (int)(i & 15);
        int mtile = (int)(tile / Kt);
        int ktile = (int)(tile % Kt);
        int half = lane >> 4, mrow = lane & 15;
        int row = mtile * 16 + mrow;
        int k = ktile * 32 + frag_klocal(e, half);
        float v = 0.0f;
        if (row < MROWS) v = emb[(long)dropidx[row] * EE + k];
        dst[i] = (__bf16)v;
    }
}

// h0 = latent @ Wd + bd into h_a f32 (rows 8..15 zero); zero h_b
__global__ void h0_kernel(const float* __restrict__ latent,
                          const float* __restrict__ Wd,
                          const float* __restrict__ bd,
                          float* __restrict__ h_a, float* __restrict__ h_b) {
    int i = blockIdx.x * blockDim.x + threadIdx.x;   // 16*1024
    int row = i >> 10, col = i & (HH - 1);
    float v = 0.0f;
    if (row < BB) {
        float acc = bd[col];
        for (int k = 0; k < EE; ++k)
            acc += latent[row * EE + k] * Wd[(long)k * HH + col];
        v = acc;
    }
    h_a[i] = v;
    h_b[i] = 0.0f;
}

// swizzle h_a f32 (16x1024) into A-fragment bf16; zero hb_bf
__global__ void h0_swz_kernel(const float* __restrict__ h,
                              __bf16* __restrict__ ha_bf,
                              __bf16* __restrict__ hb_bf) {
    int i = blockIdx.x * blockDim.x + threadIdx.x;   // 16*1024 elements
    int ktile = i >> 9;                              // single m-tile
    int lane = (i >> 4) & 31;
    int e = i & 15;
    int half = lane >> 4, mrow = lane & 15;
    int k = ktile * 32 + frag_klocal(e, half);
    ha_bf[i] = (__bf16)h[mrow * HH + k];
    hb_bf[i] = (__bf16)0.0f;
}

// GEMM on pre-swizzled operands: C(f32, MxN) = A_sw @ B_sw + bias.
// 256 threads = 8 waves; wave w -> m-tile (blockIdx.y*8+w), 4 n-tiles.
__global__ void gemm_swz_wmma_kernel(const __bf16* __restrict__ A,
                                     const __bf16* __restrict__ B,
                                     const float* __restrict__ bias,
                                     float* __restrict__ C,
                                     int M, int N, int K) {
    int tid = threadIdx.x;
    int wave = tid >> 5;
    int lane = tid & 31;
    int half = lane >> 4, lrow = lane & 15;
    int Kt = K >> 5;

    int mtile = blockIdx.y * 8 + wave;
    long ntile0 = (long)blockIdx.x * 4;

    const __bf16* Ab = A + ((long)mtile * Kt) * 512 + lane * 16;
    const __bf16* Bb = B + (ntile0 * Kt) * 512 + lane * 16;
    const long nstride = (long)Kt * 512;

    v8f c0 = {}, c1 = {}, c2 = {}, c3 = {};

    for (int kt = 0; kt < Kt; ++kt) {
        v16bf a  = *(const v16bf*)(Ab + (long)kt * 512);
        v16bf b0 = *(const v16bf*)(Bb + (long)kt * 512);
        v16bf b1 = *(const v16bf*)(Bb + (long)kt * 512 + nstride);
        v16bf b2 = *(const v16bf*)(Bb + (long)kt * 512 + 2 * nstride);
        v16bf b3 = *(const v16bf*)(Bb + (long)kt * 512 + 3 * nstride);
        if (kt + 1 < Kt)
            __builtin_prefetch(Ab + (long)(kt + 1) * 512, 0, 0);
        c0 = __builtin_amdgcn_wmma_f32_16x16x32_bf16(false, a, false, b0, (short)0, c0, false, false);
        c1 = __builtin_amdgcn_wmma_f32_16x16x32_bf16(false, a, false, b1, (short)0, c1, false, false);
        c2 = __builtin_amdgcn_wmma_f32_16x16x32_bf16(false, a, false, b2, (short)0, c2, false, false);
        c3 = __builtin_amdgcn_wmma_f32_16x16x32_bf16(false, a, false, b3, (short)0, c3, false, false);
    }

    // store: VGPR r -> M = r + half*8, N = lane&15 (per-lane coalesced)
    #pragma unroll
    for (int nt = 0; nt < 4; ++nt) {
        v8f c = (nt == 0) ? c0 : (nt == 1) ? c1 : (nt == 2) ? c2 : c3;
        long gcol = (ntile0 + nt) * 16 + lrow;
        float bv = bias ? bias[gcol] : 0.0f;
        #pragma unroll
        for (int r = 0; r < 8; ++r) {
            int grow = mtile * 16 + r + half * 8;
            if (grow < M)
                C[(long)grow * N + gcol] = c[r] + bv;
        }
    }
}

// One GRU step. 64 blocks x 32 threads; block owns 16 H-columns.
__global__ void gru_step_kernel(const __bf16* __restrict__ hbf_in,
                                const float* __restrict__ h_in,
                                float* __restrict__ h_out,
                                __bf16* __restrict__ hbf_out,
                                const __bf16* __restrict__ Uk,
                                const float* __restrict__ xproj,
                                const float* __restrict__ b1,
                                __bf16* __restrict__ hs,
                                int t) {
    int lane = threadIdx.x & 31;
    int half = lane >> 4, lrow = lane & 15;
    int j0 = blockIdx.x * 16;
    const int Kt = HH >> 5;                               // 32 k-tiles

    // B tile bases for the 3 gates (ntile = (g*1024 + j0)/16)
    long ntz = (long)(j0 >> 4);
    long ntr = (long)((HH + j0) >> 4);
    long nth = (long)((2 * HH + j0) >> 4);
    const __bf16* Bz = Uk + (ntz * Kt) * 512 + lane * 16;
    const __bf16* Br = Uk + (ntr * Kt) * 512 + lane * 16;
    const __bf16* Bh = Uk + (nth * Kt) * 512 + lane * 16;
    const __bf16* Ab = hbf_in + lane * 16;

    v8f cz = {}, cr = {}, ch = {};

    for (int kt = 0; kt < Kt; ++kt) {
        long o = (long)kt * 512;
        v16bf a  = *(const v16bf*)(Ab + o);
        v16bf bz = *(const v16bf*)(Bz + o);
        v16bf br = *(const v16bf*)(Br + o);
        v16bf bh = *(const v16bf*)(Bh + o);
        cz = __builtin_amdgcn_wmma_f32_16x16x32_bf16(false, a, false, bz, (short)0, cz, false, false);
        cr = __builtin_amdgcn_wmma_f32_16x16x32_bf16(false, a, false, br, (short)0, cr, false, false);
        ch = __builtin_amdgcn_wmma_f32_16x16x32_bf16(false, a, false, bh, (short)0, ch, false, false);
    }

    int n = j0 + lrow;                 // H column owned by this lane
    int klocal = n & 31, ktile = n >> 5;
    int half2, e2;
    inv_klocal(klocal, half2, e2);

    #pragma unroll
    for (int r = 0; r < 8; ++r) {
        int m = r + half * 8;          // batch row (0..15, valid < 8)
        if (m < BB) {
            long xb = ((long)m * TT + t) * H3 + n;
            float xz = xproj[xb];
            float xr = xproj[xb + HH];
            float xh = xproj[xb + 2 * HH];
            float rz = cz[r] + b1[n];
            float rr = cr[r] + b1[HH + n];
            float rh = ch[r] + b1[2 * HH + n];
            float z  = sigmoidf_(xz + rz);
            float rg = sigmoidf_(xr + rr);
            float hh = tanhf(xh + rg * rh);
            float hold = h_in[m * HH + n];
            float hnew = z * hold + (1.0f - z) * hh;
            h_out[m * HH + n] = hnew;
            // h bf16 fragment for next step (single m-tile)
            hbf_out[(long)ktile * 512 + (half2 * 16 + m) * 16 + e2] = (__bf16)hnew;
            // hs bf16 fragment for the logits GEMM (row = m*TT + t)
            int row = m * TT + t;
            int mtile = row >> 4, mrow = row & 15;
            hs[((long)mtile * (HH >> 5) + ktile) * 512 + (half2 * 16 + mrow) * 16 + e2] =
                (__bf16)hnew;
        }
    }
}

// ---------------- host launch ----------------
extern "C" void kernel_launch(void* const* d_in, const int* in_sizes, int n_in,
                              void* d_out, int out_size, void* d_ws, size_t ws_size,
                              hipStream_t stream) {
    const float* latent = (const float*)d_in[0];   // (8,256)
    const int*   tok    = (const int*)  d_in[1];   // (8,512)
    const float* emb    = (const float*)d_in[2];   // (32000,256)
    const float* Wd     = (const float*)d_in[3];   // (256,1024)
    const float* bd     = (const float*)d_in[4];   // (1024,)
    const float* Wk     = (const float*)d_in[5];   // (256,3072)
    const float* Uk     = (const float*)d_in[6];   // (1024,3072)
    const float* b_gru  = (const float*)d_in[7];   // (2,3072)
    const float* Wo     = (const float*)d_in[8];   // (1024,32000)
    const float* bo     = (const float*)d_in[9];   // (32000,)
    float* out = (float*)d_out;                    // (8,511,32000) f32

    char* ws = (char*)d_ws;
    __bf16* Wo_sw = (__bf16*)(ws + OFF_WO);
    __bf16* Uk_sw = (__bf16*)(ws + OFF_UK);
    __bf16* Wk_sw = (__bf16*)(ws + OFF_WK);
    __bf16* x_sw  = (__bf16*)(ws + OFF_X);
    float*  xproj = (float*) (ws + OFF_XP);
    __bf16* hs_sw = (__bf16*)(ws + OFF_HS);
    float*  h_a   = (float*) (ws + OFF_HA);
    float*  h_b   = (float*) (ws + OFF_HB);
    __bf16* ha_bf = (__bf16*)(ws + OFF_HABF);
    __bf16* hb_bf = (__bf16*)(ws + OFF_HBBF);
    int*    didx  = (int*)   (ws + OFF_IDX);

    // 1) swizzle-convert weights (B operands)
    swizzleB_kernel<<<2048, 256, 0, stream>>>(Wo, Wo_sw, HH, VV);
    swizzleB_kernel<<<1024, 256, 0, stream>>>(Uk, Uk_sw, HH, H3);
    swizzleB_kernel<<<512,  256, 0, stream>>>(Wk, Wk_sw, EE, H3);

    // 2) dropout indices, then gather+swizzle embeddings (A operand)
    dropidx_kernel<<<(MROWS + 255) / 256, 256, 0, stream>>>(tok, didx);
    swizzleA_gather_kernel<<<1024, 256, 0, stream>>>(didx, emb, x_sw);

    // 3) h0 (f32) + its bf16 fragment image
    h0_kernel<<<(16 * HH) / 256, 256, 0, stream>>>(latent, Wd, bd, h_a, h_b);
    h0_swz_kernel<<<(16 * HH) / 256, 256, 0, stream>>>(h_a, ha_bf, hb_bf);

    // 4) x_proj = x @ Wk + b_gru[0]
    {
        dim3 grid(H3 / 64, MPAD / 128);
        gemm_swz_wmma_kernel<<<grid, 256, 0, stream>>>(
            x_sw, Wk_sw, b_gru, xproj, MROWS, H3, EE);
    }

    // 5) GRU recurrence: 511 sequential steps (double-buffered h)
    for (int t = 0; t < TT; ++t) {
        const __bf16* hbi = (t & 1) ? hb_bf : ha_bf;
        __bf16*       hbo = (t & 1) ? ha_bf : hb_bf;
        const float*  hfi = (t & 1) ? h_b : h_a;
        float*        hfo = (t & 1) ? h_a : h_b;
        gru_step_kernel<<<HH / 16, 32, 0, stream>>>(
            hbi, hfi, hfo, hbo, Uk_sw, xproj, b_gru + H3, hs_sw, t);
    }

    // 6) logits = hs @ Wo + bo
    {
        dim3 grid(VV / 64, MPAD / 128);
        gemm_swz_wmma_kernel<<<grid, 256, 0, stream>>>(
            hs_sw, Wo_sw, bo, out, MROWS, VV, HH);
    }
}
